// AttentiveFuturecaster_67147518706112
// MI455X (gfx1250) — matrix operand order repaired and checked by reading, not compiled, not run on hardware
//
#include <hip/hip_runtime.h>
#include <hip/hip_bf16.h>
#include <math.h>

typedef unsigned short u16;
typedef unsigned int   u32;
typedef __attribute__((ext_vector_type(16))) __bf16 v16bf;
typedef __attribute__((ext_vector_type(8)))  __bf16 v8bf;
typedef __attribute__((ext_vector_type(8)))  float  v8f;

#define B_SZ   2048
#define T_LEN  128
#define F_IN   64
#define HDIM   256
#define GDIM   768     // 3*H
#define OUTL   32

// ---- workspace layout (bytes) ----
#define OFF_WC   0ull                         // u16[768][320]  Wih|Whh combined
#define OFF_DWHH (OFF_WC   + 768ull*320*2)    // u16[768][256]
#define OFF_FC1  (OFF_DWHH + 768ull*256*2)    // u16[256][256]
#define OFF_HS   (OFF_FC1  + 256ull*256*2)    // u16[2048][128][256]  (134MB, L2-resident)

__device__ __forceinline__ u16 f2bf(float x) {
    u32 u = __float_as_uint(x);
    u += 0x7FFFu + ((u >> 16) & 1u);
    return (u16)(u >> 16);
}
__device__ __forceinline__ float bf2f(u32 b) {
    return __uint_as_float((b & 0xFFFFu) << 16);
}

// A fragment (16x32 bf16 MxK) per ISA layout; base points at row 0 of the 16-row tile.
__device__ __forceinline__ v16bf load_a(const u16* base, int stride, int kbase, int lane) {
    int m  = lane & 15;
    int kh = kbase + ((lane >> 4) << 3);
    const u16* p0 = base + m * stride + kh;
    union { v16bf v; v8bf h[2]; } u;
    u.h[0] = *(const v8bf*)p0;
    u.h[1] = *(const v8bf*)(p0 + 16);
    return u.v;
}

// B fragment from a per-lane base + compile-time-constant element offset; the
// offset folds into the global_load_b128 instruction IOFFSET (no VALU addr math).
__device__ __forceinline__ v16bf load_b_off(const u16* base, int off) {
    union { v16bf v; v8bf h[2]; } u;
    u.h[0] = *(const v8bf*)(base + off);
    u.h[1] = *(const v8bf*)(base + off + 16);
    return u.v;
}

__device__ __forceinline__ v8f wmma_bf16(v16bf a, v16bf b, v8f c) {
    return __builtin_amdgcn_wmma_f32_16x16x32_bf16(false, a, false, b, (short)0, c, false, false);
}

__device__ __forceinline__ float sigm(float x) { return 1.f / (1.f + __expf(-x)); }

// Opaque zero: blocks LICM from hoisting weight loads across the sequential
// outer loop (hoisting previously spilled ~480 VGPRs of B-fragments to scratch).
__device__ __forceinline__ size_t opaque_zero() {
    size_t z = 0;
    asm volatile("" : "+s"(z));
    return z;
}

// ---------------- weight pack: f32 -> bf16 ----------------
__global__ void pack_kernel(const float* __restrict__ encWih, const float* __restrict__ encWhh,
                            const float* __restrict__ decWhh, const float* __restrict__ fc1W,
                            u16* __restrict__ Wc, u16* __restrict__ dWhh, u16* __restrict__ fc1) {
    int i = blockIdx.x * blockDim.x + threadIdx.x;
    if (i < 768 * 320) {
        int n = i / 320, k = i % 320;
        float v = (k < 64) ? encWih[n * 64 + k] : encWhh[n * 256 + (k - 64)];
        Wc[i] = f2bf(v);
    }
    if (i < 768 * 256) dWhh[i] = f2bf(decWhh[i]);
    if (i < 256 * 256) fc1[i]  = f2bf(fc1W[i]);
}

// ---------------- encoder: persistent per 32-row batch tile ----------------
__global__ __launch_bounds__(256) void encoder_kernel(
    const float* __restrict__ x, const float* __restrict__ h0,
    const u16* __restrict__ Wc, const float* __restrict__ bih,
    const float* __restrict__ bhh, u16* __restrict__ hs)
{
    __shared__ __attribute__((aligned(16))) u16 Ab[32][336];   // [x_t (K 0..63) | h (K 64..319)]
    __shared__ float GXS[32][776];   // gi for all gates; pass2 adds gh for r/z
    __shared__ float GHN[32][264];   // gh for n-gate only

    const int tid  = threadIdx.x;
    const int lane = tid & 31;
    const int wave = tid >> 5;
    const int b0   = blockIdx.x * 32;

    for (int i = tid; i < 32 * 256; i += 256) {
        int r = i >> 8, c = i & 255;
        Ab[r][64 + c] = f2bf(h0[(size_t)(b0 + r) * HDIM + c]);
    }
    __syncthreads();

    const int nbase = wave * 96;                       // 6 n-tiles per wave
    const int blane = (lane & 15) * 320 + ((lane >> 4) << 3);

    for (int t = 0; t < T_LEN; ++t) {
        const u16* Wct = Wc + opaque_zero();           // re-stream weights from L2 each step

        for (int i = tid; i < 32 * 64; i += 256) {
            int r = i >> 6, c = i & 63;
            Ab[r][c] = f2bf(x[((size_t)(b0 + r) * T_LEN + t) * F_IN + c]);
        }
        __syncthreads();

        // ---- pass 1: x-part gates (K = 0..63) -> GXS ----
        {
            v16bf a0[2], a1[2];
            #pragma unroll
            for (int kt = 0; kt < 2; ++kt) {
                a0[kt] = load_a(&Ab[0][0],  336, kt * 32, lane);
                a1[kt] = load_a(&Ab[16][0], 336, kt * 32, lane);
            }
            const u16* bb = Wct + (size_t)(nbase * 320 + blane);
            #pragma unroll
            for (int nt = 0; nt < 6; ++nt) {
                const int noff = nt * 16 * 320;
                v16bf bq0 = load_b_off(bb, noff);
                v16bf bq1 = load_b_off(bb, noff + 32);
                v8f c0 = {}; v8f c1 = {};
                c0 = wmma_bf16(a0[0], bq0, c0);
                c1 = wmma_bf16(a1[0], bq0, c1);
                c0 = wmma_bf16(a0[1], bq1, c0);
                c1 = wmma_bf16(a1[1], bq1, c1);
                int col = nbase + nt * 16 + (lane & 15);
                int rb  = (lane < 16) ? 0 : 8;
                #pragma unroll
                for (int r = 0; r < 8; ++r) { GXS[rb + r][col] = c0[r]; GXS[16 + rb + r][col] = c1[r]; }
            }
        }

        // ---- pass 2: h-part gates (K = 64..319); same wave/lane owns the same
        // (row,col) as pass 1, so the += below is ordered by per-wave DS ordering.
        {
            v16bf a0[8], a1[8];
            #pragma unroll
            for (int kt = 0; kt < 8; ++kt) {
                a0[kt] = load_a(&Ab[0][0],  336, 64 + kt * 32, lane);
                a1[kt] = load_a(&Ab[16][0], 336, 64 + kt * 32, lane);
            }
            const u16* bb = Wct + (size_t)(nbase * 320 + blane + 64);
            #pragma unroll
            for (int nt = 0; nt < 6; ++nt) {
                const int n    = nbase + nt * 16;
                const int noff = nt * 16 * 320;
                v8f c0 = {}; v8f c1 = {};
                v16bf bcur = load_b_off(bb, noff);            // double-buffered B stream
                #pragma unroll
                for (int kt = 0; kt < 8; ++kt) {
                    v16bf bnxt = bcur;
                    if (kt < 7) bnxt = load_b_off(bb, noff + (kt + 1) * 32);
                    c0 = wmma_bf16(a0[kt], bcur, c0);
                    c1 = wmma_bf16(a1[kt], bcur, c1);
                    bcur = bnxt;
                }
                int col = n + (lane & 15);
                int rb  = (lane < 16) ? 0 : 8;
                if (n < 512) {   // r,z gates: only gi+gh sum is needed
                    #pragma unroll
                    for (int r = 0; r < 8; ++r) { GXS[rb + r][col] += c0[r]; GXS[16 + rb + r][col] += c1[r]; }
                } else {         // n gate: keep gh separate (r scales only gh)
                    #pragma unroll
                    for (int r = 0; r < 8; ++r) { GHN[rb + r][col - 512] = c0[r]; GHN[16 + rb + r][col - 512] = c1[r]; }
                }
            }
        }
        __syncthreads();

        // ---- GRU combine ----
        {
            int r  = tid >> 3;
            int jb = (tid & 7) << 5;
            #pragma unroll 8
            for (int jj = 0; jj < 32; ++jj) {
                int j = jb + jj;
                float rg = sigm(GXS[r][j] + bih[j] + bhh[j]);
                float zg = sigm(GXS[r][256 + j] + bih[256 + j] + bhh[256 + j]);
                float ng = tanhf(GXS[r][512 + j] + bih[512 + j] + rg * (GHN[r][j] + bhh[512 + j]));
                float hv = (1.f - zg) * ng + zg * bf2f(Ab[r][64 + j]);
                u16 hb = f2bf(hv);
                Ab[r][64 + j] = hb;
                hs[(((size_t)(b0 + r)) * T_LEN + t) * HDIM + j] = hb;
            }
        }
        __syncthreads();
    }
}

// ---------------- decoder: persistent per 16-row batch tile ----------------
__global__ __launch_bounds__(256) void decoder_kernel(
    const float* __restrict__ x, const u16* __restrict__ hs,
    const u16* __restrict__ dWhh, const float* __restrict__ dWih,
    const float* __restrict__ dbih, const float* __restrict__ dbhh,
    const float* __restrict__ Wq, const float* __restrict__ bq,
    const u16* __restrict__ fc1w, const float* __restrict__ fc1b,
    const float* __restrict__ fc2w, const float* __restrict__ fc2b,
    float* __restrict__ out)
{
    __shared__ float Qs[16][256];
    __shared__ float Sc[16][128];
    __shared__ __attribute__((aligned(16))) u16 Cx[16][272];
    __shared__ float GHs[16][776];
    __shared__ __attribute__((aligned(16))) u16 Hd[16][272];
    __shared__ float F1[16][256];
    __shared__ float Prev[16];

    const int tid  = threadIdx.x;
    const int lane = tid & 31;
    const int wave = tid >> 5;
    const int b0   = blockIdx.x * 16;

    if (tid < 16) Prev[tid] = x[((size_t)(b0 + tid) * T_LEN + (T_LEN - 1)) * F_IN + 0];
    __syncthreads();

    const float scale = 0.0625f;   // 1/sqrt(256)
    const int blane = (lane & 15) * 256 + ((lane >> 4) << 3);
    union U8 { uint4 q; u16 s[8]; };

    for (int s = 0; s < OUTL; ++s) {
        const u16* dWhh_t = dWhh + opaque_zero();
        const u16* fc1w_t = fc1w + opaque_zero();

        // q = prev*Wq + bq
        for (int i = tid; i < 16 * 256; i += 256) {
            int r = i >> 8, c = i & 255;
            Qs[r][c] = Prev[r] * Wq[c] + bq[c];
        }
        __syncthreads();

        // scores[b][t] = q[b].hs[b][t] * scale   (hs bf16, served from L2)
        for (int p = tid; p < 16 * 128; p += 256) {
            int r = p >> 7, tt = p & 127;
            const u16* hp = hs + (((size_t)(b0 + r)) * T_LEN + tt) * HDIM;
            float acc = 0.f;
            for (int j = 0; j < 256; j += 8) {
                U8 a; a.q = *(const uint4*)(hp + j);
                #pragma unroll
                for (int u = 0; u < 8; ++u) acc += Qs[r][j + u] * bf2f(a.s[u]);
            }
            Sc[r][tt] = acc * scale;
        }
        __syncthreads();

        // softmax over t
        if (tid < 16) {
            float mx = -1e30f;
            for (int tt = 0; tt < 128; ++tt) mx = fmaxf(mx, Sc[tid][tt]);
            float sum = 0.f;
            for (int tt = 0; tt < 128; ++tt) { float e = __expf(Sc[tid][tt] - mx); Sc[tid][tt] = e; sum += e; }
            float inv = 1.f / sum;
            for (int tt = 0; tt < 128; ++tt) Sc[tid][tt] *= inv;
        }
        __syncthreads();

        // ctx[b] = sum_t w[t]*hs[b][t]
        {
            int r = tid >> 4, jb = (tid & 15) << 4;
            float acc[16];
            #pragma unroll
            for (int u = 0; u < 16; ++u) acc[u] = 0.f;
            for (int tt = 0; tt < 128; ++tt) {
                float w = Sc[r][tt];
                const u16* hp = hs + (((size_t)(b0 + r)) * T_LEN + tt) * HDIM + jb;
                U8 a0, a1; a0.q = *(const uint4*)hp; a1.q = *(const uint4*)(hp + 8);
                #pragma unroll
                for (int u = 0; u < 8; ++u) { acc[u] += w * bf2f(a0.s[u]); acc[8 + u] += w * bf2f(a1.s[u]); }
            }
            #pragma unroll
            for (int u = 0; u < 16; ++u) Cx[r][jb + u] = f2bf(acc[u]);
        }
        __syncthreads();

        // gh = ctx @ dec_Whh.T   (16x256)@(256x768) via WMMA
        {
            v16bf afr[8];
            #pragma unroll
            for (int kt = 0; kt < 8; ++kt) afr[kt] = load_a(&Cx[0][0], 272, kt * 32, lane);
            const u16* bb = dWhh_t + (size_t)(wave * 96 * 256 + blane);
            #pragma unroll
            for (int nt = 0; nt < 6; ++nt) {
                const int noff = nt * 16 * 256;
                v8f acc = {};
                v16bf bcur = load_b_off(bb, noff);
                #pragma unroll
                for (int kt = 0; kt < 8; ++kt) {
                    v16bf bnxt = bcur;
                    if (kt < 7) bnxt = load_b_off(bb, noff + (kt + 1) * 32);
                    acc = wmma_bf16(afr[kt], bcur, acc);
                    bcur = bnxt;
                }
                int col = wave * 96 + nt * 16 + (lane & 15);
                int rb  = (lane < 16) ? 0 : 8;
                #pragma unroll
                for (int r = 0; r < 8; ++r) GHs[rb + r][col] = acc[r];
            }
        }
        __syncthreads();

        // GRU combine (x-part rank-1 in prev); hd = (1-z)*n + z*ctx
        {
            int r = tid >> 4, jb = (tid & 15) << 4;
            float pv = Prev[r];
            #pragma unroll
            for (int jj = 0; jj < 16; ++jj) {
                int j = jb + jj;
                float rg = sigm(pv * dWih[j] + dbih[j] + GHs[r][j] + dbhh[j]);
                float zg = sigm(pv * dWih[256 + j] + dbih[256 + j] + GHs[r][256 + j] + dbhh[256 + j]);
                float ng = tanhf(pv * dWih[512 + j] + dbih[512 + j] + rg * (GHs[r][512 + j] + dbhh[512 + j]));
                float hd = (1.f - zg) * ng + zg * bf2f(Cx[r][j]);
                Hd[r][j] = f2bf(hd);
            }
        }
        __syncthreads();

        // fc1: relu(hd @ fc1_W.T + b)   (16x256)@(256x256) via WMMA
        {
            v16bf afr[8];
            #pragma unroll
            for (int kt = 0; kt < 8; ++kt) afr[kt] = load_a(&Hd[0][0], 272, kt * 32, lane);
            const u16* bb = fc1w_t + (size_t)(wave * 32 * 256 + blane);
            #pragma unroll
            for (int nt = 0; nt < 2; ++nt) {
                const int noff = nt * 16 * 256;
                v8f acc = {};
                v16bf bcur = load_b_off(bb, noff);
                #pragma unroll
                for (int kt = 0; kt < 8; ++kt) {
                    v16bf bnxt = bcur;
                    if (kt < 7) bnxt = load_b_off(bb, noff + (kt + 1) * 32);
                    acc = wmma_bf16(afr[kt], bcur, acc);
                    bcur = bnxt;
                }
                int col = wave * 32 + nt * 16 + (lane & 15);
                int rb  = (lane < 16) ? 0 : 8;
                #pragma unroll
                for (int r = 0; r < 8; ++r) F1[rb + r][col] = fmaxf(acc[r] + fc1b[col], 0.f);
            }
        }
        __syncthreads();

        // fc2 + output + carry prev
        if (tid < 16) {
            float acc = fc2b[0];
            for (int j = 0; j < 256; ++j) acc += F1[tid][j] * fc2w[j];
            out[(size_t)(b0 + tid) * OUTL + s] = acc;
            Prev[tid] = acc;
        }
        __syncthreads();
    }
}

extern "C" void kernel_launch(void* const* d_in, const int* in_sizes, int n_in,
                              void* d_out, int out_size, void* d_ws, size_t ws_size,
                              hipStream_t stream) {
    const float* x       = (const float*)d_in[0];
    const float* h0      = (const float*)d_in[1];
    const float* encWih  = (const float*)d_in[2];
    const float* encWhh  = (const float*)d_in[3];
    const float* enc_bih = (const float*)d_in[4];
    const float* enc_bhh = (const float*)d_in[5];
    const float* decWih  = (const float*)d_in[6];
    const float* decWhh  = (const float*)d_in[7];
    const float* dec_bih = (const float*)d_in[8];
    const float* dec_bhh = (const float*)d_in[9];
    const float* attnWq  = (const float*)d_in[10];
    const float* attnbq  = (const float*)d_in[11];
    const float* fc1W    = (const float*)d_in[12];
    const float* fc1b    = (const float*)d_in[13];
    const float* fc2W    = (const float*)d_in[14];
    const float* fc2b    = (const float*)d_in[15];

    u16* Wc   = (u16*)((char*)d_ws + OFF_WC);
    u16* dWhh = (u16*)((char*)d_ws + OFF_DWHH);
    u16* fc1w = (u16*)((char*)d_ws + OFF_FC1);
    u16* hs   = (u16*)((char*)d_ws + OFF_HS);

    pack_kernel<<<(768 * 320 + 255) / 256, 256, 0, stream>>>(
        encWih, encWhh, decWhh, fc1W, Wc, dWhh, fc1w);

    encoder_kernel<<<B_SZ / 32, 256, 0, stream>>>(
        x, h0, Wc, enc_bih, enc_bhh, hs);

    decoder_kernel<<<B_SZ / 16, 256, 0, stream>>>(
        x, hs, dWhh, decWih, dec_bih, dec_bhh, attnWq, attnbq,
        fc1w, fc1b, fc2W, fc2b, (float*)d_out);
}
